// SE_GNN_73340861546734
// MI455X (gfx1250) — matrix operand order repaired and checked by reading, not compiled, not run on hardware
//
#include <hip/hip_runtime.h>
#include <hip/hip_bf16.h>
#include <math.h>

#define N_ENT   1559
#define N_LAYER 4
#define HDIM    128
#define NEDGE   300000
#define BQ      4096
#define EPS_BN  1e-5f

#define MTILES   ((N_ENT + 15) / 16)        // 98
#define MGROUPS  ((MTILES + 3) / 4)         // 25  (4 m-tiles per block)
#define M_PAD    (MGROUPS * 4 * 16)         // 1600 padded rows
#define NTILES   (HDIM / 16)                // 8
#define BSTRIDE  136                        // padded k-stride (272 B) to stagger LDS banks

typedef __attribute__((ext_vector_type(16))) _Float16 v16h;
typedef __attribute__((ext_vector_type(8)))  _Float16 v8h;
typedef __attribute__((ext_vector_type(8)))  float    v8f;

// ---- monotone uint encoding of float for atomicMax-based segment_max ----
__device__ __forceinline__ unsigned flip_f32(float f) {
    unsigned u = __float_as_uint(f);
    return (u & 0x80000000u) ? ~u : (u | 0x80000000u);
}
__device__ __forceinline__ float unflip_f32(unsigned u) {
    return __uint_as_float((u & 0x80000000u) ? (u & 0x7FFFFFFFu) : ~u);
}

// ---------------------------------------------------------------------------
// Zero accumulators: agg[N_ENT*H], z[N_ENT], mflip[N_ENT]
// ---------------------------------------------------------------------------
__global__ void zero_accum_kernel(float* __restrict__ agg,
                                  float* __restrict__ z,
                                  unsigned* __restrict__ mflip) {
    int i = blockIdx.x * blockDim.x + threadIdx.x;
    if (i < N_ENT * HDIM) agg[i] = 0.0f;
    if (i < N_ENT) { z[i] = 0.0f; mflip[i] = 0u; }   // 0 < flip(any real)
}

__global__ void copy_kernel(const float* __restrict__ src, float* __restrict__ dst, int n) {
    int i = blockIdx.x * blockDim.x + threadIdx.x;
    if (i < n) dst[i] = src[i];
}

// Pre-convert a weight tensor (f32 row-major) to f16 once per launch.
__global__ void cvt_f16_kernel(const float* __restrict__ src, _Float16* __restrict__ dst, int n) {
    int i = blockIdx.x * blockDim.x + threadIdx.x;
    if (i < n) dst[i] = (_Float16)src[i];
}

// ---------------------------------------------------------------------------
// Per-edge message (per-lane float4, lane covers features [4*lane, 4*lane+4) )
//   mode 0 (Edge): msg = r          score = r . e_dst
//   mode 1 (Node): msg = e_src      score = e_src . e_dst
//   mode 2 (Comp): msg = e_src * r  score = (e_src*r) . e_dst
// ---------------------------------------------------------------------------
__device__ __forceinline__ float4 edge_msg(const float* __restrict__ ent,
                                           const float* __restrict__ rel,
                                           int s, int r, int lane, int mode) {
    if (mode == 0) {
        return ((const float4*)(rel + (size_t)r * HDIM))[lane];
    } else if (mode == 1) {
        return ((const float4*)(ent + (size_t)s * HDIM))[lane];
    } else {
        float4 a = ((const float4*)(ent + (size_t)s * HDIM))[lane];
        float4 b = ((const float4*)(rel + (size_t)r * HDIM))[lane];
        return make_float4(a.x * b.x, a.y * b.y, a.z * b.z, a.w * b.w);
    }
}

// Pass 1: score[e] = msg . e_dst ; segment max over dst (flip-encoded atomicMax)
__global__ void edge_score_max_kernel(const float* __restrict__ ent,
                                      const float* __restrict__ rel,
                                      const int* __restrict__ src,
                                      const int* __restrict__ dst,
                                      const int* __restrict__ rid,
                                      float* __restrict__ score,
                                      unsigned* __restrict__ mflip,
                                      int mode) {
    int wave = blockIdx.x * (blockDim.x >> 5) + (threadIdx.x >> 5);
    int lane = threadIdx.x & 31;
    if (wave >= NEDGE) return;
    int d = dst[wave];
    float4 dv = ((const float4*)(ent + (size_t)d * HDIM))[lane];
    float4 mv = edge_msg(ent, rel, src[wave], rid[wave], lane, mode);
    float p = mv.x * dv.x + mv.y * dv.y + mv.z * dv.z + mv.w * dv.w;
#pragma unroll
    for (int off = 16; off > 0; off >>= 1) p += __shfl_xor(p, off, 32);
    if (lane == 0) {
        score[wave] = p;
        atomicMax(mflip + d, flip_f32(p));
    }
}

// Pass 2: w = exp(score - m[dst]); z[dst] += w; agg[dst] += msg * w
__global__ void edge_accum_kernel(const float* __restrict__ ent,
                                  const float* __restrict__ rel,
                                  const int* __restrict__ src,
                                  const int* __restrict__ dst,
                                  const int* __restrict__ rid,
                                  const float* __restrict__ score,
                                  const unsigned* __restrict__ mflip,
                                  float* __restrict__ z,
                                  float* __restrict__ agg,
                                  int mode) {
    int wave = blockIdx.x * (blockDim.x >> 5) + (threadIdx.x >> 5);
    int lane = threadIdx.x & 31;
    if (wave >= NEDGE) return;
    int d = dst[wave];
    float* arow = agg + (size_t)d * HDIM + lane * 4;
    __builtin_prefetch(arow, 1, 1);                    // global_prefetch of scatter target
    float m = unflip_f32(mflip[d]);
    float w = __expf(score[wave] - m);
    if (lane == 0) atomicAdd(z + d, w);
    float4 mv = edge_msg(ent, rel, src[wave], rid[wave], lane, mode);
    atomicAdd(arow + 0, mv.x * w);
    atomicAdd(arow + 1, mv.y * w);
    atomicAdd(arow + 2, mv.z * w);
    atomicAdd(arow + 3, mv.w * w);
}

// Pass 3: softmax-normalize and emit padded f16 matrix for the WMMA GEMM.
__global__ void normalize_f16_kernel(const float* __restrict__ agg,
                                     const float* __restrict__ z,
                                     _Float16* __restrict__ agg_h) {
    int i = blockIdx.x * blockDim.x + threadIdx.x;
    if (i >= M_PAD * HDIM) return;
    int node = i >> 7;                                  // / HDIM
    float v = 0.0f;
    if (node < N_ENT) {
        float zz = z[node];
        v = (zz > 0.0f) ? agg[i] / zz : 0.0f;
    }
    agg_h[i] = (_Float16)v;
}

// ---------------------------------------------------------------------------
// WMMA GEMM: H = A[M_PAD x 128](f16) @ W[128 x 128](f16) -> f32 [M_PAD x 128]
// Block = 4 waves; each wave owns one 16x16 output tile (4 consecutive m-tiles
// per block, fixed n-tile). The B panel is staged in LDS *transposed* into the
// fragment layout Bt[n][k] (k-stride padded to 136 halves to stagger banks),
// so every fragment build is two aligned ds_load_b128s. K fully unrolled.
// Hbuf is padded to M_PAD rows so the C store is unconditional (no execz
// ladder); downstream kernels read only rows < N_ENT.
// A/B fragments use contiguous k = 16*half + e (same in both); C/D layout per CDNA5 ISA 7.12.2.
// ---------------------------------------------------------------------------
__global__ void gemm_wmma_kernel(const _Float16* __restrict__ Ah,
                                 const _Float16* __restrict__ Wh,
                                 float* __restrict__ Hbuf) {
    __shared__ _Float16 Bt[16 * BSTRIDE];               // 4.25 KB transposed B panel

    int nt  = blockIdx.x % NTILES;
    int mg  = blockIdx.x / NTILES;
    int tid = threadIdx.x;                              // 0..127  (= k row during fill)

    // Cooperative transpose-fill: thread t reads W[t][n0..n0+15] (two b128),
    // scatters 16 halves to Bt[n][t].
    {
        const v8h* wrow = (const v8h*)(Wh + (size_t)tid * HDIM + nt * 16);
        v8h w0 = wrow[0];
        v8h w1 = wrow[1];
#pragma unroll
        for (int n = 0; n < 8; ++n) {
            Bt[n * BSTRIDE + tid]       = w0[n];
            Bt[(n + 8) * BSTRIDE + tid] = w1[n];
        }
    }
    __syncthreads();

    int wave = tid >> 5;
    int lane = tid & 31;
    int half = lane >> 4;                               // lanes 0-15 vs 16-31
    int l16  = lane & 15;
    int mt   = mg * 4 + wave;
    int row  = mt * 16 + l16;                           // A row held by this lane (< M_PAD)

    const v8h* arow = (const v8h*)(Ah + (size_t)row * HDIM);
    const _Float16* bcol = Bt + l16 * BSTRIDE;          // this lane's B column
    v8f c = {};
#pragma unroll
    for (int ks = 0; ks < 4; ++ks) {                    // kt = 32*ks
        // A frag: element e -> K = kt + 16*half + e
        v8h alo = arow[ks * 4 + 2 * half];
        v8h ahi = arow[ks * 4 + 2 * half + 1];
        v16h a = __builtin_shufflevector(alo, ahi, 0, 1, 2, 3, 4, 5, 6, 7,
                                         8, 9, 10, 11, 12, 13, 14, 15);
        // B frag: element e -> K = kt + 16*half + e : 16 contiguous halves in Bt
        const v8h* bk = (const v8h*)(bcol + ks * 32 + 16 * half);
        v8h blo = bk[0];
        v8h bhi = bk[1];
        v16h b = __builtin_shufflevector(blo, bhi, 0, 1, 2, 3, 4, 5, 6, 7,
                                         8, 9, 10, 11, 12, 13, 14, 15);
        c = __builtin_amdgcn_wmma_f32_16x16x32_f16(false, a, false, b,
                                                   (short)0, c, false, false);
    }
    // Unconditional store into the M_PAD-tall padded output (f32 C/D layout:
    // VGPR v -> row mt*16 + v + 8*half, column nt*16 + l16).
    float* hout = Hbuf + (size_t)(mt * 16 + (half ? 8 : 0)) * HDIM + nt * 16 + l16;
#pragma unroll
    for (int v = 0; v < 8; ++v)
        hout[(size_t)v * HDIM] = c[v];
}

// Column batch stats over N_ENT rows (biased variance, matching reference)
__global__ void colstats_kernel(const float* __restrict__ Hbuf,
                                float* __restrict__ mu, float* __restrict__ var) {
    __shared__ float ssum[256];
    __shared__ float ssq[256];
    int col = blockIdx.x;
    int tid = threadIdx.x;
    float s = 0.0f, q = 0.0f;
    for (int r = tid; r < N_ENT; r += blockDim.x) {
        float v = Hbuf[(size_t)r * HDIM + col];
        s += v; q += v * v;
    }
    ssum[tid] = s; ssq[tid] = q;
    __syncthreads();
    for (int off = 128; off > 0; off >>= 1) {
        if (tid < off) { ssum[tid] += ssum[tid + off]; ssq[tid] += ssq[tid + off]; }
        __syncthreads();
    }
    if (tid == 0) {
        float m = ssum[0] / (float)N_ENT;
        mu[col]  = m;
        var[col] = ssq[0] / (float)N_ENT - m * m;
    }
}

// ent_next += tanh((h - mu) * rsqrt(var + eps) * g + b)
__global__ void bn_tanh_add_kernel(const float* __restrict__ Hbuf,
                                   const float* __restrict__ mu,
                                   const float* __restrict__ var,
                                   const float* __restrict__ g,
                                   const float* __restrict__ b,
                                   float* __restrict__ ent_next) {
    int i = blockIdx.x * blockDim.x + threadIdx.x;
    if (i >= N_ENT * HDIM) return;
    int col = i & (HDIM - 1);
    float x = (Hbuf[i] - mu[col]) * rsqrtf(var[col] + EPS_BN) * g[col] + b[col];
    ent_next[i] += tanhf(x);
}

// out = concat(ent[drug1], ent[drug2])
__global__ void gather_kernel(const float* __restrict__ ent,
                              const int* __restrict__ d1,
                              const int* __restrict__ d2,
                              float* __restrict__ out) {
    int i = blockIdx.x * blockDim.x + threadIdx.x;
    if (i >= 2 * BQ * HDIM) return;
    int part = i / (BQ * HDIM);
    int j = i - part * BQ * HDIM;
    int bq = j / HDIM, h = j - bq * HDIM;
    int node = part ? d2[bq] : d1[bq];
    out[i] = ent[(size_t)node * HDIM + h];
}

// ---------------------------------------------------------------------------
extern "C" void kernel_launch(void* const* d_in, const int* in_sizes, int n_in,
                              void* d_out, int out_size, void* d_ws, size_t ws_size,
                              hipStream_t stream) {
    const float* ent_emb  = (const float*)d_in[0];
    const float* rel_emb  = (const float*)d_in[1];
    const float* W3[3]  = { (const float*)d_in[2], (const float*)d_in[3], (const float*)d_in[4] };
    const float* G3[3]  = { (const float*)d_in[5], (const float*)d_in[7], (const float*)d_in[9] };
    const float* Bb3[3] = { (const float*)d_in[6], (const float*)d_in[8], (const float*)d_in[10] };
    const int* src = (const int*)d_in[11];
    const int* dst = (const int*)d_in[12];
    const int* rid = (const int*)d_in[13];
    const int* d1  = (const int*)d_in[14];
    const int* d2  = (const int*)d_in[15];
    float* out = (float*)d_out;

    // Workspace carve-up (256B aligned chunks)
    char* p = (char*)d_ws;
    auto carve = [&](size_t bytes) { char* q = p; p += (bytes + 255) & ~(size_t)255; return q; };
    float*    ent_a  = (float*)carve(sizeof(float) * N_ENT * HDIM);
    float*    ent_b  = (float*)carve(sizeof(float) * N_ENT * HDIM);
    float*    agg    = (float*)carve(sizeof(float) * N_ENT * HDIM);
    _Float16* agg_h  = (_Float16*)carve(sizeof(_Float16) * M_PAD * HDIM);
    float*    hbuf   = (float*)carve(sizeof(float) * M_PAD * HDIM);   // padded rows
    _Float16* wh     = (_Float16*)carve(sizeof(_Float16) * 3 * N_LAYER * HDIM * HDIM);
    float*    score  = (float*)carve(sizeof(float) * NEDGE);
    float*    z      = (float*)carve(sizeof(float) * N_ENT);
    unsigned* mflip  = (unsigned*)carve(sizeof(unsigned) * N_ENT);
    float*    mu     = (float*)carve(sizeof(float) * HDIM);
    float*    var    = (float*)carve(sizeof(float) * HDIM);

    const int elemBlocks = (N_ENT * HDIM + 255) / 256;           // node*feature grids
    const int padBlocks  = (M_PAD * HDIM + 255) / 256;
    const int edgeBlocks = (NEDGE + 7) / 8;                      // 8 waves / 256-thr block
    const int gemmBlocks = NTILES * MGROUPS;                     // 200 blocks, 128 thr

    // One-time per launch: weights -> f16 (3 tensors x [4,128,128])
    for (int t = 0; t < 3; ++t) {
        cvt_f16_kernel<<<(N_LAYER * HDIM * HDIM + 255) / 256, 256, 0, stream>>>(
            W3[t], wh + (size_t)t * N_LAYER * HDIM * HDIM, N_LAYER * HDIM * HDIM);
    }

    copy_kernel<<<elemBlocks, 256, 0, stream>>>(ent_emb, ent_a, N_ENT * HDIM);

    float* cur = ent_a;
    float* nxt = ent_b;
    for (int l = 0; l < N_LAYER; ++l) {
        copy_kernel<<<elemBlocks, 256, 0, stream>>>(cur, nxt, N_ENT * HDIM);
        for (int t = 0; t < 3; ++t) {
            zero_accum_kernel<<<elemBlocks, 256, 0, stream>>>(agg, z, mflip);
            edge_score_max_kernel<<<edgeBlocks, 256, 0, stream>>>(cur, rel_emb, src, dst, rid,
                                                                  score, mflip, t);
            edge_accum_kernel<<<edgeBlocks, 256, 0, stream>>>(cur, rel_emb, src, dst, rid,
                                                              score, mflip, z, agg, t);
            normalize_f16_kernel<<<padBlocks, 256, 0, stream>>>(agg, z, agg_h);
            gemm_wmma_kernel<<<gemmBlocks, 128, 0, stream>>>(
                agg_h, wh + (size_t)(t * N_LAYER + l) * HDIM * HDIM, hbuf);
            colstats_kernel<<<HDIM, 256, 0, stream>>>(hbuf, mu, var);
            bn_tanh_add_kernel<<<elemBlocks, 256, 0, stream>>>(hbuf, mu, var,
                                                               G3[t] + (size_t)l * HDIM,
                                                               Bb3[t] + (size_t)l * HDIM, nxt);
        }
        float* tmp = cur; cur = nxt; nxt = tmp;
    }

    gather_kernel<<<(2 * BQ * HDIM + 255) / 256, 256, 0, stream>>>(cur, d1, d2, out);
}
